// LLoCaAttention_21492016349800
// MI455X (gfx1250) — compile-verified, matrix operand order checked by reading, not run on hardware
//
#include <hip/hip_runtime.h>
#include <hip/hip_bf16.h>

#define B_ 4
#define H_ 8
#define N_ 2048
#define C_ 32

typedef __attribute__((ext_vector_type(16))) __bf16 v16bf;
typedef __attribute__((ext_vector_type(8)))  float  v8f;

using bf16 = __hip_bfloat16;

__device__ __forceinline__ unsigned short f2bf(float f) {
    __hip_bfloat16 h = __float2bfloat16(f);
    return *reinterpret_cast<unsigned short*>(&h);
}
__device__ __forceinline__ unsigned pack2(float a, float b) {
    return (unsigned)f2bf(a) | ((unsigned)f2bf(b) << 16);
}
// xor-16 lane exchange (swap 16-lane halves): ds_swizzle group-of-32 encoding
__device__ __forceinline__ float xor16(float x) {
    return __int_as_float(__builtin_amdgcn_ds_swizzle(__float_as_int(x), 0x401F));
}
// raw v_exp_f32 (base-2). Scores are pre-scaled by log2e so exp2 == exp.
__device__ __forceinline__ float fexp2(float x) {
    return __builtin_amdgcn_exp2f(x);
}

// ---------------------------------------------------------------------------
// Kernel 1: frame transforms -> bf16 q_g (row-major, pre-scaled by log2e/C),
//           bf16 k_g (row-major), bf16 v_g transposed to [bh][c][n].
// ---------------------------------------------------------------------------
__global__ __launch_bounds__(256) void lloca_transform_kernel(
    const float* __restrict__ q, const float* __restrict__ k,
    const float* __restrict__ v, const float* __restrict__ frames,
    bf16* __restrict__ qg, bf16* __restrict__ kg, bf16* __restrict__ vgT)
{
    const int flat = blockIdx.x * blockDim.x + threadIdx.x;   // (b,h,n) flat
    const int n  = flat % N_;
    const int bh = flat / N_;
    const int b  = bh / H_;
    const size_t tok = (size_t)bh * N_ + n;

    float F[16];
    const float4* fp = (const float4*)(frames + ((size_t)b * N_ + n) * 16);
    #pragma unroll
    for (int i = 0; i < 4; ++i) {
        float4 t = fp[i];
        F[4*i+0] = t.x; F[4*i+1] = t.y; F[4*i+2] = t.z; F[4*i+3] = t.w;
    }
    const float eta[4] = {1.f, -1.f, -1.f, -1.f};
    const float qs = 1.4426950408889634f / (float)C_;   // log2(e)/C

    float xq[C_], xk[C_], xv[C_];
    {
        const float4* qr = (const float4*)(q + tok * C_);
        const float4* kr = (const float4*)(k + tok * C_);
        const float4* vr = (const float4*)(v + tok * C_);
        #pragma unroll
        for (int i = 0; i < 8; ++i) {
            float4 a = qr[i]; xq[4*i]=a.x; xq[4*i+1]=a.y; xq[4*i+2]=a.z; xq[4*i+3]=a.w;
            float4 c = kr[i]; xk[4*i]=c.x; xk[4*i+1]=c.y; xk[4*i+2]=c.z; xk[4*i+3]=c.w;
            float4 d = vr[i]; xv[4*i]=d.x; xv[4*i+1]=d.y; xv[4*i+2]=d.z; xv[4*i+3]=d.w;
        }
    }

    float oq[C_], ok[C_], ov[C_];
    #pragma unroll
    for (int c = 0; c < 16; ++c) { oq[c] = xq[c] * qs; ok[c] = xk[c]; ov[c] = xv[c]; }
    #pragma unroll
    for (int vv = 0; vv < 4; ++vv) {
        const int base = 16 + 4 * vv;
        #pragma unroll
        for (int i = 0; i < 4; ++i) {
            // inv[i][j] = F[j][i]*eta_i*eta_j ; lower_inv[i][j] = F[j][i]
            float aq = 0.f, ak = 0.f, av = 0.f;
            #pragma unroll
            for (int j = 0; j < 4; ++j) {
                const float Fji = F[4*j + i];
                aq += Fji * eta[j] * xq[base + j];
                ak += Fji *          xk[base + j];
                av += Fji * eta[j] * xv[base + j];
            }
            oq[base + i] = eta[i] * aq * qs;
            ok[base + i] = ak;
            ov[base + i] = eta[i] * av;
        }
    }

    {
        unsigned wq[16], wk[16];
        #pragma unroll
        for (int i = 0; i < 16; ++i) {
            wq[i] = pack2(oq[2*i], oq[2*i+1]);
            wk[i] = pack2(ok[2*i], ok[2*i+1]);
        }
        uint4* dq = (uint4*)(qg + tok * C_);
        uint4* dk = (uint4*)(kg + tok * C_);
        #pragma unroll
        for (int i = 0; i < 4; ++i) {
            dq[i] = make_uint4(wq[4*i], wq[4*i+1], wq[4*i+2], wq[4*i+3]);
            dk[i] = make_uint4(wk[4*i], wk[4*i+1], wk[4*i+2], wk[4*i+3]);
        }
    }
    {
        bf16* vb = vgT + (size_t)bh * C_ * N_ + n;
        #pragma unroll
        for (int c = 0; c < C_; ++c)
            vb[(size_t)c * N_] = __float2bfloat16(ov[c]);
    }
}

// ---------------------------------------------------------------------------
// Kernel 2: transposed flash attention. One wave = 32 queries (two 16-wide
// query sets sharing the K/V fragments); k-blocks of 32.
//   S^T = K * Q^T   (queries are lane columns -> in-lane softmax, exp2-domain)
//   O^T = V^T * P^T (P^T D-layout == B-fragment layout: no LDS staging)
// Per k-block: 8 global_load_b128 feed 8 WMMAs. No LDS anywhere.
// ---------------------------------------------------------------------------
__global__ __launch_bounds__(256) void lloca_attn_kernel(
    const bf16* __restrict__ qg, const bf16* __restrict__ kg,
    const bf16* __restrict__ vgT, const float* __restrict__ frames,
    float* __restrict__ out)
{
    const int wave = threadIdx.x >> 5;
    const int lane = threadIdx.x & 31;
    const int wgid = blockIdx.x * 8 + wave;     // 0 .. 2047
    const int bh   = wgid >> 6;                 // 64 q-tiles (of 32) per (b,h)
    const int qt   = wgid & 63;
    const int b    = bh / H_;
    const int q0   = qt * 32;

    const int m  = lane & 15;                   // query column / matrix row
    const int hi = lane >> 4;                   // K-chunk select per ISA layout

    // ---- Q^T B-fragments: lane = query column, values = channel chunks
    v16bf bq0, bq1;
    {
        const bf16* qrow0 = qg + ((size_t)bh * N_ + q0 + m) * C_;
        *(uint4*)&bq0       = *(const uint4*)(qrow0 + 8 * hi);
        *((uint4*)&bq0 + 1) = *(const uint4*)(qrow0 + 16 + 8 * hi);
        const bf16* qrow1 = qrow0 + 16 * C_;
        *(uint4*)&bq1       = *(const uint4*)(qrow1 + 8 * hi);
        *((uint4*)&bq1 + 1) = *(const uint4*)(qrow1 + 16 + 8 * hi);
    }

    v8f o00 = {}, o01 = {};                     // query set 0: c rows 0-15 / 16-31
    v8f o10 = {}, o11 = {};                     // query set 1
    float m0 = -3.0e38f, l0 = 0.f;              // per-query state (per-lane scalar)
    float m1 = -3.0e38f, l1 = 0.f;

    const bf16* kbh   = kg  + (size_t)bh * N_ * C_;
    const bf16* vrow0 = vgT + ((size_t)bh * C_ + m)      * N_;  // c = m
    const bf16* vrow1 = vgT + ((size_t)bh * C_ + 16 + m) * N_;  // c = 16+m

    for (int k0 = 0; k0 < N_; k0 += 32) {
        // K A-fragments: lane = key row, values = channel chunks (row-major kg)
        v16bf ak0, ak1, av0, av1;
        {
            const bf16* kr0 = kbh + (size_t)(k0 + m) * C_;
            *(uint4*)&ak0       = *(const uint4*)(kr0 + 8 * hi);
            *((uint4*)&ak0 + 1) = *(const uint4*)(kr0 + 16 + 8 * hi);
            const bf16* kr1 = kbh + (size_t)(k0 + 16 + m) * C_;
            *(uint4*)&ak1       = *(const uint4*)(kr1 + 8 * hi);
            *((uint4*)&ak1 + 1) = *(const uint4*)(kr1 + 16 + 8 * hi);
            // V^T A-fragments: lane = channel row, values = contiguous key chunks
            *(uint4*)&av0       = *(const uint4*)(vrow0 + k0 + 8 * hi);
            *((uint4*)&av0 + 1) = *(const uint4*)(vrow0 + k0 + 16 + 8 * hi);
            *(uint4*)&av1       = *(const uint4*)(vrow1 + k0 + 8 * hi);
            *((uint4*)&av1 + 1) = *(const uint4*)(vrow1 + k0 + 16 + 8 * hi);
        }

        #pragma unroll
        for (int qs = 0; qs < 2; ++qs) {
            const v16bf& bq = qs ? bq1 : bq0;
            v8f& oa = qs ? o10 : o00;
            v8f& ob = qs ? o11 : o01;
            float& mi = qs ? m1 : m0;
            float& li = qs ? l1 : l0;

            // S^T tiles (keys x queries), fp32 accum; scores already in exp2-domain
            v8f s0 = {}, s1 = {};
            s0 = __builtin_amdgcn_wmma_f32_16x16x32_bf16(false, ak0, false, bq, (short)0, s0, false, false);
            s1 = __builtin_amdgcn_wmma_f32_16x16x32_bf16(false, ak1, false, bq, (short)0, s1, false, false);

            // online softmax, per query == per lane column (16 of 32 keys here,
            // the other 16 live in lane^16)
            float mx = fmaxf(s0[0], s1[0]);
            #pragma unroll
            for (int r = 1; r < 8; ++r) mx = fmaxf(mx, fmaxf(s0[r], s1[r]));
            mx = fmaxf(mx, xor16(mx));
            const float mnew  = fmaxf(mi, mx);
            const float alpha = fexp2(mi - mnew);
            mi = mnew;

            float p0[8], p1[8], rs = 0.f;
            #pragma unroll
            for (int r = 0; r < 8; ++r) {
                p0[r] = fexp2(s0[r] - mnew);
                p1[r] = fexp2(s1[r] - mnew);
                rs += p0[r] + p1[r];
            }
            rs += xor16(rs);
            li = li * alpha + rs;

            #pragma unroll
            for (int r = 0; r < 8; ++r) { oa[r] *= alpha; ob[r] *= alpha; }

            // P^T B-fragment built in registers:
            //   elements 0..7  = keys 8hi+0..7    (s0 rows)
            //   elements 8..15 = keys 16+8hi+0..7 (s1 rows)
            v16bf bp;
            {
                unsigned* pw = (unsigned*)&bp;
                #pragma unroll
                for (int r = 0; r < 4; ++r) pw[r]     = pack2(p0[2*r], p0[2*r+1]);
                #pragma unroll
                for (int r = 0; r < 4; ++r) pw[4 + r] = pack2(p1[2*r], p1[2*r+1]);
            }

            oa = __builtin_amdgcn_wmma_f32_16x16x32_bf16(false, av0, false, bp, (short)0, oa, false, false);
            ob = __builtin_amdgcn_wmma_f32_16x16x32_bf16(false, av1, false, bp, (short)0, ob, false, false);
        }
    }

    // ---- epilogue (fully in-lane, no LDS), once per query set:
    // lane holds query q, channels c = 8hi+r (oa) and 16+8hi+r (ob).
    #pragma unroll
    for (int qs = 0; qs < 2; ++qs) {
        v8f oa = qs ? o10 : o00;
        v8f ob = qs ? o11 : o01;
        const float li  = qs ? l1 : l0;
        const float inl = 1.0f / li;
        #pragma unroll
        for (int r = 0; r < 8; ++r) { oa[r] *= inl; ob[r] *= inl; }

        const int n = q0 + 16 * qs + m;
        float F[16];
        {
            const float4* fp = (const float4*)(frames + ((size_t)b * N_ + n) * 16);
            #pragma unroll
            for (int i = 0; i < 4; ++i) {
                float4 t = fp[i];
                F[4*i+0] = t.x; F[4*i+1] = t.y; F[4*i+2] = t.z; F[4*i+3] = t.w;
            }
        }
        // ob holds two complete 4-vectors in consecutive elements
        float vo[8];
        #pragma unroll
        for (int vb = 0; vb < 2; ++vb) {
            #pragma unroll
            for (int i = 0; i < 4; ++i) {
                float acc = 0.f;
                #pragma unroll
                for (int j = 0; j < 4; ++j) acc += F[4*i + j] * ob[4*vb + j];
                vo[4*vb + i] = acc;
            }
        }

        float* rowp = out + ((size_t)bh * N_ + n) * C_;
        float4* sp = (float4*)(rowp + 8 * hi);       // scalar channels c = 8hi..+7
        sp[0] = make_float4(oa[0], oa[1], oa[2], oa[3]);
        sp[1] = make_float4(oa[4], oa[5], oa[6], oa[7]);
        float4* vp = (float4*)(rowp + 16 + 8 * hi);  // vector channels c = 16+8hi..+7
        vp[0] = make_float4(vo[0], vo[1], vo[2], vo[3]);
        vp[1] = make_float4(vo[4], vo[5], vo[6], vo[7]);
    }
}

// ---------------------------------------------------------------------------
extern "C" void kernel_launch(void* const* d_in, const int* in_sizes, int n_in,
                              void* d_out, int out_size, void* d_ws, size_t ws_size,
                              hipStream_t stream) {
    const float* q      = (const float*)d_in[0];
    const float* k      = (const float*)d_in[1];
    const float* v      = (const float*)d_in[2];
    const float* frames = (const float*)d_in[3];
    float* out = (float*)d_out;

    const size_t elems = (size_t)B_ * H_ * N_ * C_;
    bf16* qg  = (bf16*)d_ws;
    bf16* kg  = qg + elems;
    bf16* vgT = kg + elems;

    lloca_transform_kernel<<<(B_ * H_ * N_) / 256, 256, 0, stream>>>(
        q, k, v, frames, qg, kg, vgT);
    lloca_attn_kernel<<<(B_ * H_ * N_ / 32) / 8, 256, 0, stream>>>(
        qg, kg, vgT, frames, out);
}
